// LoopABMIL_87419764343315
// MI455X (gfx1250) — matrix operand order repaired and checked by reading, not compiled, not run on hardware
//
#include <hip/hip_runtime.h>
#include <math.h>

typedef __attribute__((ext_vector_type(16))) __bf16 v16bf;
typedef __attribute__((ext_vector_type(8)))  float  v8f;

#define BB 8
#define NN 8192
#define DD 1024
#define HH 256
#define CC 2
#define MM (BB*NN)        // 65536 rows total
#define KTILES (DD/32)    // 32 K-steps of 32
#define NTILES (HH/16)    // 16 N-tiles of 16
#define BTILE_BF16 (NTILES*32*16)   // one kt's B fragments: 8192 bf16 = 16 KB

// ---------------------------------------------------------------------------
// Kernel 1: repack Wp (f32 [D,H]) into bf16 WMMA B-fragments.
// Fragment layout used here (16-bit B 32x16, wave32, contiguous K order):
//   lane l holds column n = nt*16 + (l&15), K values kb+e with kb=(l>=16)?16:0
// Storage: wb[ ((kt*16+nt)*32 + lane)*16 + e ]  -> 512 KB total, L2-resident.
// ---------------------------------------------------------------------------
__global__ void repack_wp(const float* __restrict__ Wp, __bf16* __restrict__ wb) {
    int frag = blockIdx.x;            // kt*16 + nt
    int kt = frag >> 4, nt = frag & 15;
    int l  = threadIdx.x;             // 0..31
    int col = nt*16 + (l & 15);
    int kb  = (l >> 4) * 16;
    __bf16* o = wb + ((size_t)frag*32 + l)*16;
#pragma unroll
    for (int e = 0; e < 16; ++e) {
        int K = kt*32 + kb + e;
        o[e] = (__bf16)Wp[(size_t)K*HH + col];
    }
}

// ---------------------------------------------------------------------------
// Kernel 2: h = silu(x @ Wp + bp) via v_wmma_f32_16x16x32_bf16, fused with
// a = h @ Wa + ba.
//
// One wave per 16-row M-tile, full H=256 in 16 v8f accumulators so x is read
// exactly once from HBM (256 MB ~ 11us at 23.3 TB/s = roofline floor).
// B tiles (16 KB per kt) are staged in LDS with GLOBAL_LOAD_ASYNC_TO_LDS_B128
// (ASYNCcnt), double-buffered, shared by all 8 waves of the block -> cuts the
// B-side L2 traffic 8x (2 GB -> 256 MB). A fragments software-pipelined one
// K-step ahead.
// ---------------------------------------------------------------------------
__global__ void __launch_bounds__(256)
gemm_silu_attn(const float* __restrict__ x, const __bf16* __restrict__ wb,
               const float* __restrict__ bp, const float* __restrict__ wa,
               const float* __restrict__ ba,
               float* __restrict__ h, float* __restrict__ aout)
{
    __shared__ __attribute__((aligned(128))) __bf16 btile[2][BTILE_BF16]; // 2 x 16 KB

    const int tid   = threadIdx.x;
    const int lane  = tid & 31;
    const int wave  = tid >> 5;
    const int tile  = blockIdx.x * 8 + wave;   // 0..4095
    const int mbase = tile * 16;
    const int lm    = lane & 15;
    const int hi    = lane >> 4;               // 0 or 1 (half-wave)
    const float* xr = x + (size_t)(mbase + lm) * DD;
    const int q     = hi * 4;                  // float4 index offset for A frag

    // per-thread source/dest for the cooperative async B-tile copy
    const unsigned long long gB = (unsigned long long)(uintptr_t)wb + (unsigned)tid * 16u;
    const unsigned ldsB0 = (unsigned)(uintptr_t)&btile[0][0] + (unsigned)tid * 16u;
    const unsigned ldsB1 = (unsigned)(uintptr_t)&btile[1][0] + (unsigned)tid * 16u;

    // prefetch B tile for kt into LDS buffer buf (4 x b128 per thread = 16 KB/block)
    auto prefetchB = [&](int kt, int buf) {
        unsigned long long g = gB + (unsigned long long)kt * (BTILE_BF16 * 2);
        unsigned l = buf ? ldsB1 : ldsB0;
#pragma unroll
        for (int r = 0; r < 4; ++r) {
            asm volatile("global_load_async_to_lds_b128 %0, %1, off"
                         :: "v"(l + (unsigned)(r * 4096)),
                            "v"(g + (unsigned long long)(r * 4096))
                         : "memory");
        }
    };

    v8f acc[NTILES];
#pragma unroll
    for (int t = 0; t < NTILES; ++t) acc[t] = (v8f){0.f,0.f,0.f,0.f,0.f,0.f,0.f,0.f};

    prefetchB(0, 0);                           // ASYNCcnt = 4

    // preload A fragment source for kt = 0
    float4 acur0, acur1, acur2, acur3;
    {
        const float4* x4 = (const float4*)xr;
        acur0 = x4[q+0]; acur1 = x4[q+1]; acur2 = x4[q+2]; acur3 = x4[q+3];
    }

#pragma unroll 1
    for (int kt = 0; kt < KTILES; ++kt) {
        const bool more = (kt + 1 < KTILES);
        if (more) prefetchB(kt + 1, (kt + 1) & 1);   // ASYNCcnt -> 8

        // issue next A-fragment loads early (overlap with this step's compute)
        float4 anext0, anext1, anext2, anext3;
        if (more) {
            const float4* x4 = (const float4*)(xr + (kt + 1) * 32);
            anext0 = x4[q+0]; anext1 = x4[q+1]; anext2 = x4[q+2]; anext3 = x4[q+3];
        }

        // wait until tile kt's 4 async loads have landed (in-order completion),
        // leaving tile kt+1's 4 still in flight; then make LDS visible block-wide.
        if (more) asm volatile("s_wait_asynccnt 4" ::: "memory");
        else      asm volatile("s_wait_asynccnt 0" ::: "memory");
        __syncthreads();

        // A fragment (16-bit A 16x32, contiguous K order used by this code):
        //  row = lane&15; K=kbase+e, kbase=(lane>=16)?16:0 (matches packed B)
        v16bf af;
        af[0] =(__bf16)acur0.x; af[1] =(__bf16)acur0.y; af[2] =(__bf16)acur0.z; af[3] =(__bf16)acur0.w;
        af[4] =(__bf16)acur1.x; af[5] =(__bf16)acur1.y; af[6] =(__bf16)acur1.z; af[7] =(__bf16)acur1.w;
        af[8] =(__bf16)acur2.x; af[9] =(__bf16)acur2.y; af[10]=(__bf16)acur2.z; af[11]=(__bf16)acur2.w;
        af[12]=(__bf16)acur3.x; af[13]=(__bf16)acur3.y; af[14]=(__bf16)acur3.z; af[15]=(__bf16)acur3.w;

        const v16bf* bl = (const v16bf*)&btile[kt & 1][(size_t)lane * 16];
#pragma unroll
        for (int nt = 0; nt < NTILES; ++nt) {
            v16bf bf = bl[nt * 32];            // fragment stride = 32 lanes * 32B = 1 KB
            acc[nt] = __builtin_amdgcn_wmma_f32_16x16x32_bf16(
                false, af, false, bf, (short)0, acc[nt], false, false);
        }

        __syncthreads();                       // protect buffer (kt&1) before overwrite
        acur0 = anext0; acur1 = anext1; acur2 = anext2; acur3 = anext3;
    }

    // Epilogue: bias + SiLU, store h, fused attention-logit partial (h . Wa).
    // C/D layout: VGPR r, lanes 0-15 -> (M=r, N=lane), lanes 16-31 -> (M=r+8).
    const float ba0 = ba[0];
    float part[8];
#pragma unroll
    for (int r = 0; r < 8; ++r) part[r] = 0.f;

#pragma unroll
    for (int nt = 0; nt < NTILES; ++nt) {
        int col = nt*16 + lm;
        float bpc = bp[col];
        float wac = wa[col];
        float* hp = h + (size_t)(mbase + hi*8)*HH + col;
#pragma unroll
        for (int r = 0; r < 8; ++r) {
            float v = acc[nt][r] + bpc;
            float s = v / (1.f + __expf(-v));   // silu
            hp[(size_t)r*HH] = s;
            part[r] += s * wac;
        }
    }
    // Reduce across the 16 lanes of each half-wave (xor of bits 0..3 stays in half).
#pragma unroll
    for (int r = 0; r < 8; ++r) {
        float t = part[r];
        t += __shfl_xor(t, 1, 32);
        t += __shfl_xor(t, 2, 32);
        t += __shfl_xor(t, 4, 32);
        t += __shfl_xor(t, 8, 32);
        part[r] = t;
    }
    if (lm == 0) {
#pragma unroll
        for (int r = 0; r < 8; ++r)
            aout[mbase + hi*8 + r] = part[r] + ba0;
    }
}

// ---------------------------------------------------------------------------
// Kernel 3: per-bag masked softmax over a[b, 0..len).  One block per bag.
// ---------------------------------------------------------------------------
__global__ void __launch_bounds__(1024)
softmax_ragged(const float* __restrict__ a, const int* __restrict__ lengths,
               float* __restrict__ p)
{
    __shared__ float red[32];
    __shared__ float bval[2];
    const int b   = blockIdx.x;
    const int len = lengths[b];
    const float* ab = a + (size_t)b*NN;
    float* pb       = p + (size_t)b*NN;
    const int tid = threadIdx.x, lane = tid & 31, wv = tid >> 5;

    float m = -INFINITY;
    for (int i = tid; i < NN; i += 1024)
        if (i < len) m = fmaxf(m, ab[i]);
#pragma unroll
    for (int s = 16; s > 0; s >>= 1) m = fmaxf(m, __shfl_xor(m, s, 32));
    if (lane == 0) red[wv] = m;
    __syncthreads();
    if (wv == 0) {
        float t = red[lane];
#pragma unroll
        for (int s = 16; s > 0; s >>= 1) t = fmaxf(t, __shfl_xor(t, s, 32));
        if (lane == 0) bval[0] = t;
    }
    __syncthreads();
    m = bval[0];

    float sum = 0.f;
    for (int i = tid; i < NN; i += 1024)
        if (i < len) sum += __expf(ab[i] - m);
#pragma unroll
    for (int s = 16; s > 0; s >>= 1) sum += __shfl_xor(sum, s, 32);
    if (lane == 0) red[wv] = sum;
    __syncthreads();
    if (wv == 0) {
        float t = red[lane];
#pragma unroll
        for (int s = 16; s > 0; s >>= 1) t += __shfl_xor(t, s, 32);
        if (lane == 0) bval[1] = t;
    }
    __syncthreads();
    const float inv = 1.f / bval[1];
    for (int i = tid; i < NN; i += 1024)
        pb[i] = (i < len) ? __expf(ab[i] - m) * inv : 0.f;
}

// ---------------------------------------------------------------------------
// Kernel 4: pooled[b,:] = sum_n p[b,n]*h[b,n,:]; logits = pooled@Wc + bc.
// One block per bag; p staged in LDS; fixed-order tree reduction (deterministic).
// ---------------------------------------------------------------------------
__global__ void __launch_bounds__(256)
pool_classify(const float* __restrict__ h, const float* __restrict__ p,
              const float* __restrict__ Wc, const float* __restrict__ bc,
              float* __restrict__ out)
{
    __shared__ float sp[NN];            // 32 KB of 320 KB LDS
    __shared__ float r0[256], r1[256];
    const int b = blockIdx.x, c = threadIdx.x;
    for (int i = c; i < NN; i += 256) sp[i] = p[(size_t)b*NN + i];
    __syncthreads();

    const float* hb = h + (size_t)b*NN*HH + c;
    float acc = 0.f;
#pragma unroll 4
    for (int n = 0; n < NN; ++n)
        acc += sp[n] * hb[(size_t)n*HH];

    r0[c] = acc * Wc[c*CC + 0];
    r1[c] = acc * Wc[c*CC + 1];
    __syncthreads();
    for (int s = 128; s > 0; s >>= 1) {
        if (c < s) { r0[c] += r0[c+s]; r1[c] += r1[c+s]; }
        __syncthreads();
    }
    if (c == 0) {
        out[b*CC + 0] = r0[0] + bc[0];
        out[b*CC + 1] = r1[0] + bc[1];
    }
}

// ---------------------------------------------------------------------------
extern "C" void kernel_launch(void* const* d_in, const int* in_sizes, int n_in,
                              void* d_out, int out_size, void* d_ws, size_t ws_size,
                              hipStream_t stream)
{
    const float* x       = (const float*)d_in[0];
    const int*   lengths = (const int*)  d_in[1];
    const float* Wp      = (const float*)d_in[2];
    const float* bp      = (const float*)d_in[3];
    const float* Wa      = (const float*)d_in[4];
    const float* ba      = (const float*)d_in[5];
    const float* Wc      = (const float*)d_in[6];
    const float* bc      = (const float*)d_in[7];
    float* out = (float*)d_out;

    // Workspace layout:
    //   [0, 512KB)            packed bf16 Wp fragments
    //   [1MB, 1MB+64MB)       h  (f32, B*N*H)
    //   [+64MB, +256KB)       a  (f32, B*N)
    //   [+256KB, +256KB)      p  (f32, B*N)
    char* ws = (char*)d_ws;
    __bf16* wb  = (__bf16*)(ws);
    float* hbuf = (float*)(ws + (1u << 20));
    float* abuf = (float*)(ws + (1u << 20) + (size_t)MM*HH*4);
    float* pbuf = (float*)(ws + (1u << 20) + (size_t)MM*HH*4 + (size_t)MM*4);

    repack_wp     <<<KTILES*NTILES, 32,   0, stream>>>(Wp, wb);
    gemm_silu_attn<<<MM/16/8,       256,  0, stream>>>(x, wb, bp, Wa, ba, hbuf, abuf);
    softmax_ragged<<<BB,            1024, 0, stream>>>(abuf, lengths, pbuf);
    pool_classify <<<BB,            256,  0, stream>>>(hbuf, pbuf, Wc, bc, out);
}
